// LSTMAnomalyDetector_65962107732315
// MI455X (gfx1250) — matrix operand order repaired and checked by reading, not compiled, not run on hardware
//
#include <hip/hip_runtime.h>

#define B_  4096
#define T_  512
#define I_  3
#define H_  64
#define G_  256   // 4*H

typedef _Float16 v16h __attribute__((ext_vector_type(16)));
typedef _Float16 v8h  __attribute__((ext_vector_type(8)));
typedef float    v8f  __attribute__((ext_vector_type(8)));

__device__ __forceinline__ float fsig(float xv) {
    // sigmoid(x) = 1 / (1 + exp(-x)), exp via native exp2
    float e = __builtin_amdgcn_exp2f(-1.442695040889f * xv);
    return __builtin_amdgcn_rcpf(1.0f + e);
}
__device__ __forceinline__ float ftanh(float xv) {
    // tanh(x) = 2/(1+exp(-2x)) - 1
    float e = __builtin_amdgcn_exp2f(-2.885390081777f * xv);
    return fmaf(2.0f, __builtin_amdgcn_rcpf(1.0f + e), -1.0f);
}

__global__ __launch_bounds__(128)
void lstm_fused_kernel(const float* __restrict__ x,
                       const float* __restrict__ W_ih,
                       const float* __restrict__ W_hh,
                       const float* __restrict__ b_ih,
                       const float* __restrict__ b_hh,
                       const float* __restrict__ b_dec,
                       const float* __restrict__ W_dec,
                       float* __restrict__ out)
{
    __shared__ _Float16 hlds[16 * H_];   // h(t) tile: 16 rows x 64 cols, f16
    __shared__ float    wdec[I_ * H_];   // decoder weights

    const int tid  = threadIdx.x;
    const int wave = tid >> 5;           // 0..3 : owns h columns [16*wave, 16*wave+16)
    const int lane = tid & 31;
    const int lh   = lane >> 4;          // lane half (0/1)
    const int lm   = lane & 15;
    const int rowBase = blockIdx.x * 16; // 16 batch rows per block

    for (int i = tid; i < 16 * H_; i += 128) hlds[i] = (_Float16)0.0f;
    for (int i = tid; i < I_ * H_; i += 128) wdec[i] = W_dec[i];

    // ---- time-invariant per-lane constants ----
    // W_hh B-fragments (f16, kept resident in VGPRs) for this wave's 4 gate tiles.
    // B layout (32x16 f16): col=lm; lane<16 -> elems = k 0..7,16..23 ; lane>=16 -> k 8..15,24..31.
    v16h wb[4][2];
    float wx0[4], wx1[4], wx2[4], bs[4];
    #pragma unroll
    for (int q = 0; q < 4; ++q) {                 // q = gate (i,f,g,o)
        const int g = 64 * q + 16 * wave + lm;    // gate column in [0,256)
        #pragma unroll
        for (int fr = 0; fr < 2; ++fr) {
            #pragma unroll
            for (int e = 0; e < 16; ++e) {
                const int k = 32 * fr + 8 * lh + (e < 8 ? e : 8 + e);
                wb[q][fr][e] = (_Float16)W_hh[g * H_ + k];
            }
        }
        wx0[q] = W_ih[g * I_ + 0];
        wx1[q] = W_ih[g * I_ + 1];
        wx2[q] = W_ih[g * I_ + 2];
        bs[q]  = b_ih[g] + b_hh[g];
    }

    // fused decoder: threads 0..47 each own one (row, out-channel) dot product
    const bool dec_on = (tid < 48);
    const int  drow = tid / 3;
    const int  dcol = tid - 3 * drow;
    const float bdec = dec_on ? b_dec[dcol] : 0.0f;

    v8f cst;                                  // cell state tile (f32, registers)
    #pragma unroll
    for (int r = 0; r < 8; ++r) cst[r] = 0.0f;

    __syncthreads();

    for (int t = 0; t < T_; ++t) {
        // x values for this lane's 8 rows (lanes 0-15: rows 0..7, 16-31: rows 8..15)
        float xa[8], xb[8], xc[8];
        {
            const float* xp = x + ((size_t)(rowBase + lh * 8) * T_ + t) * I_;
            #pragma unroll
            for (int r = 0; r < 8; ++r) {
                const float* p = xp + (size_t)r * (T_ * I_);
                xa[r] = p[0]; xb[r] = p[1]; xc[r] = p[2];
            }
        }

        __syncthreads();  // hlds now holds complete h(t-1)

        // A fragments of h(t-1): 16x32 f16, two K-halves
        v16h a0, a1;
        {
            const _Float16* rp = &hlds[lm * H_];
            v8h l0 = *(const v8h*)(rp +  0 + lh * 8);
            v8h h0 = *(const v8h*)(rp + 16 + lh * 8);
            v8h l1 = *(const v8h*)(rp + 32 + lh * 8);
            v8h h1 = *(const v8h*)(rp + 48 + lh * 8);
            #pragma unroll
            for (int e = 0; e < 8; ++e) {
                a0[e] = l0[e]; a0[e + 8] = h0[e];
                a1[e] = l1[e]; a1[e + 8] = h1[e];
            }
        }

        // fused decode of h(t-1) -> out[:, t-1, :]
        if (t > 0 && dec_on) {
            float acc = bdec;
            #pragma unroll 8
            for (int k = 0; k < H_; ++k)
                acc = fmaf((float)hlds[drow * H_ + k], wdec[dcol * H_ + k], acc);
            out[((size_t)(rowBase + drow) * T_ + (t - 1)) * I_ + dcol] = acc;
        }

        // gates: C = x*W_ih + b, then D = A x B + C via WMMA (K=64 in 2 steps)
        v8f d[4];
        #pragma unroll
        for (int q = 0; q < 4; ++q) {
            v8f c;
            #pragma unroll
            for (int r = 0; r < 8; ++r)
                c[r] = fmaf(xa[r], wx0[q],
                        fmaf(xb[r], wx1[q],
                          fmaf(xc[r], wx2[q], bs[q])));
            c = __builtin_amdgcn_wmma_f32_16x16x32_f16(false, a1, false, wb[q][1],
                                                       (short)0, c, false, false);
            c = __builtin_amdgcn_wmma_f32_16x16x32_f16(false, a0, false, wb[q][0],
                                                       (short)0, c, false, false);
            d[q] = c;
        }

        // activations + state update (PyTorch gate order i,f,g,o)
        float hval[8];
        #pragma unroll
        for (int r = 0; r < 8; ++r) {
            float ig = fsig(d[0][r]);
            float fg = fsig(d[1][r]);
            float gg = ftanh(d[2][r]);
            float og = fsig(d[3][r]);
            float cn = fmaf(fg, cst[r], ig * gg);
            cst[r]  = cn;
            hval[r] = og * ftanh(cn);
        }

        __syncthreads();  // all reads of h(t-1) from hlds are done

        // write h(t) tile (this wave's 16 columns)
        {
            const int col = 16 * wave + lm;
            #pragma unroll
            for (int r = 0; r < 8; ++r)
                hlds[(lh * 8 + r) * H_ + col] = (_Float16)hval[r];
        }
    }

    __syncthreads();
    // epilogue: decode final timestep h(T-1)
    if (dec_on) {
        float acc = bdec;
        #pragma unroll 8
        for (int k = 0; k < H_; ++k)
            acc = fmaf((float)hlds[drow * H_ + k], wdec[dcol * H_ + k], acc);
        out[((size_t)(rowBase + drow) * T_ + (T_ - 1)) * I_ + dcol] = acc;
    }
}

extern "C" void kernel_launch(void* const* d_in, const int* in_sizes, int n_in,
                              void* d_out, int out_size, void* d_ws, size_t ws_size,
                              hipStream_t stream) {
    (void)in_sizes; (void)n_in; (void)d_ws; (void)ws_size; (void)out_size;
    const float* x     = (const float*)d_in[0];
    const float* W_ih  = (const float*)d_in[1];
    const float* W_hh  = (const float*)d_in[2];
    const float* b_ih  = (const float*)d_in[3];
    const float* b_hh  = (const float*)d_in[4];
    const float* W_dec = (const float*)d_in[5];
    const float* b_dec = (const float*)d_in[6];
    float* out = (float*)d_out;

    dim3 grid(B_ / 16);   // 256 blocks, one 16-row batch tile each
    dim3 block(128);      // 4 waves: one gate-quadruple per wave
    lstm_fused_kernel<<<grid, block, 0, stream>>>(x, W_ih, W_hh, b_ih, b_hh,
                                                  b_dec, W_dec, out);
}
